// ConvPixelInceptionConvDecoder_2Layer_27023934226677
// MI455X (gfx1250) — matrix operand order repaired and checked by reading, not compiled, not run on hardware
//
#include <hip/hip_runtime.h>
#include <hip/hip_bf16.h>

#define NTOT 110592            // 48*48*48
#define NCHUNK 54              // N / 2048

typedef __attribute__((ext_vector_type(16))) __bf16 bf16x16;
typedef __attribute__((ext_vector_type(8)))  float  f32x8;

union ABFrag { unsigned u[8]; bf16x16 v; };

static __device__ __forceinline__ unsigned short f2bf(float f) {
    __bf16 h = (__bf16)f;
    return __builtin_bit_cast(unsigned short, h);
}
static __device__ __forceinline__ unsigned pack2(float lo, float hi) {
    return (unsigned)f2bf(lo) | ((unsigned)f2bf(hi) << 16);
}
// A-matrix (16x32 bf16) K offset for VGPR i, lane-half hi: V0-3 -> K0..7 / K8..15(hi), V4-7 -> K16..23 / K24..31(hi)
static __device__ __forceinline__ int ka(int i, int hi) {
    return ((i < 4) ? 0 : 16) + hi * 8 + (i & 3) * 2;
}
// B-matrix (32x16 bf16): same K order as A: V0-3 -> K0..7 / K8..15(hi), V4-7 -> K16..23 / K24..31(hi)
static __device__ __forceinline__ int kb(int i, int hi) {
    return ((i < 4) ? 0 : 16) + hi * 8 + (i & 3) * 2;
}

// ---------------- Kernel 1: q/k projection GEMM  W[128x256] @ x[b][256xN] ----------------
// grid (N/64, 2 row-halves, B), block 256 (8 waves). 64KB LDS.
__global__ __launch_bounds__(256) void proj_kernel(const float* __restrict__ x,
                                                   const float* __restrict__ W,
                                                   float* __restrict__ outp) {
    __shared__ unsigned short sW[64 * 256];
    __shared__ unsigned short sX[64 * 256];
    const int t  = threadIdx.x;
    const int n0 = blockIdx.x * 64;
    const int rh = blockIdx.y * 64;
    const int b  = blockIdx.z;

    // weights rows [rh, rh+64): 8192 bf16-pairs
    const float* Wb = W + rh * 256;
    for (int i = 0; i < 32; ++i) {
        int p = i * 256 + t;
        float2 wv = *(const float2*)(Wb + 2 * p);
        *(unsigned*)&sW[2 * p] = pack2(wv.x, wv.y);
    }
    // x tile: sX[col*256 + c], c contiguous (pairs packed)
    for (int i = 0; i < 32; ++i) {
        int p   = i * 256 + t;         // [0, 8192)
        int col = p & 63;
        int c0  = (p >> 6) * 2;
        const float* xp = x + (size_t)(b * 256 + c0) * NTOT + n0 + col;
        *(unsigned*)&sX[col * 256 + c0] = pack2(xp[0], xp[NTOT]);
    }
    __syncthreads();

    const int l  = t & 31;
    const int wv = t >> 5;
    const int mt = wv & 3;
    const int nt0 = (wv >> 2) * 2;
    const int lm = l & 15;
    const int hi = (l >= 16) ? 1 : 0;

    f32x8 acc0 = {}; f32x8 acc1 = {};
    const int m  = mt * 16 + lm;
    const int c0 = nt0 * 16 + lm;
    for (int k0 = 0; k0 < 256; k0 += 32) {
        ABFrag a, b0, b1;
#pragma unroll
        for (int i = 0; i < 8; ++i) {
            a.u[i]  = *(const unsigned*)&sW[m * 256 + k0 + ka(i, hi)];
            int kbo = k0 + kb(i, hi);
            b0.u[i] = *(const unsigned*)&sX[c0 * 256 + kbo];
            b1.u[i] = *(const unsigned*)&sX[(c0 + 16) * 256 + kbo];
        }
        acc0 = __builtin_amdgcn_wmma_f32_16x16x32_bf16(false, a.v, false, b0.v, (short)0, acc0, false, false);
        acc1 = __builtin_amdgcn_wmma_f32_16x16x32_bf16(false, a.v, false, b1.v, (short)0, acc1, false, false);
    }
#pragma unroll
    for (int r = 0; r < 8; ++r) {
        int ch   = rh + mt * 16 + r + hi * 8;
        int col0 = n0 + nt0 * 16 + lm;
        outp[(size_t)(b * 128 + ch) * NTOT + col0]      = acc0[r];
        outp[(size_t)(b * 128 + ch) * NTOT + col0 + 16] = acc1[r];
    }
}

// ---------------- Kernel 2: per-row online max / sum-exp for k softmax over N ----------------
__global__ __launch_bounds__(256) void kstats_kernel(const float* __restrict__ klog,
                                                     float* __restrict__ kmax,
                                                     float* __restrict__ ksum) {
    __shared__ float sm[256], ss[256];
    const int row = blockIdx.x;         // b*128 + r
    const int t   = threadIdx.x;
    const float* p = klog + (size_t)row * NTOT;
    float m = -3.4e38f, s = 0.f;
    for (int n = t; n < NTOT; n += 256) {
        float v = p[n];
        if (v > m) { s *= __expf(m - v); m = v; }
        s += __expf(v - m);
    }
    sm[t] = m; ss[t] = s;
    __syncthreads();
    for (int ofs = 128; ofs > 0; ofs >>= 1) {
        if (t < ofs) {
            float m2 = sm[t + ofs], s2 = ss[t + ofs];
            float mm = fmaxf(sm[t], m2);
            ss[t] = ss[t] * __expf(sm[t] - mm) + s2 * __expf(m2 - mm);
            sm[t] = mm;
        }
        __syncthreads();
    }
    if (t == 0) { kmax[row] = sm[0]; ksum[row] = ss[0]; }
}

// ---------------- Kernel 3: depthwise 3x3x3 conv (SAME) -> bf16 v ----------------
// grid (216 tiles, C, B), block 256. 8x8x8 outputs per tile, 10^3 halo in LDS.
__global__ __launch_bounds__(256) void vconv_kernel(const float* __restrict__ x,
                                                    const float* __restrict__ Wv,
                                                    const float* __restrict__ bv,
                                                    unsigned short* __restrict__ vbuf) {
    __shared__ float sh[1000];
    __shared__ float sw[27];
    __shared__ float sb;
    const int t = threadIdx.x;
    const int tile = blockIdx.x;
    const int c = blockIdx.y, b = blockIdx.z;
    const int tz = tile % 6, ty = (tile / 6) % 6, tx = tile / 36;
    const int h0 = tx * 8, w0 = ty * 8, d0 = tz * 8;
    if (t < 27) sw[t] = Wv[c * 27 + t];
    if (t == 31) sb = bv[c];
    const float* xb = x + (size_t)(b * 256 + c) * NTOT;
    for (int i = t; i < 1000; i += 256) {
        int hh = i / 100, rem = i % 100, ww = rem / 10, dd = rem % 10;
        int gh = h0 + hh - 1, gw = w0 + ww - 1, gd = d0 + dd - 1;
        float v = 0.f;
        if ((unsigned)gh < 48u && (unsigned)gw < 48u && (unsigned)gd < 48u)
            v = xb[(gh * 48 + gw) * 48 + gd];
        sh[i] = v;
    }
    __syncthreads();
#pragma unroll
    for (int j = 0; j < 2; ++j) {
        int o  = t + j * 256;
        int oh = o >> 6, ow = (o >> 3) & 7, od = o & 7;
        float acc = sb;
#pragma unroll
        for (int dh = 0; dh < 3; ++dh)
#pragma unroll
            for (int dw = 0; dw < 3; ++dw)
#pragma unroll
                for (int dd = 0; dd < 3; ++dd)
                    acc += sw[(dh * 3 + dw) * 3 + dd] * sh[(oh + dh) * 100 + (ow + dw) * 10 + od + dd];
        vbuf[(size_t)(b * 256 + c) * NTOT + ((h0 + oh) * 48 + (w0 + ow)) * 48 + (d0 + od)] = f2bf(acc);
    }
}

// ---------------- Kernel 4: kv partials:  softmax_N(k)[32 x N] @ v^T[N x 64] per (b,h) ----------------
// grid (54 chunks, 4 heads, B), block 256 (8 waves). 48KB LDS. Deterministic partials (no atomics).
__global__ __launch_bounds__(256) void kv_kernel(const float* __restrict__ klog,
                                                 const float* __restrict__ kmax,
                                                 const float* __restrict__ ksum,
                                                 const unsigned short* __restrict__ vbuf,
                                                 float* __restrict__ partial) {
    __shared__ unsigned short sE[32 * 256];
    __shared__ unsigned short sV[64 * 256];
    const int t = threadIdx.x;
    const int chunk = blockIdx.x, h = blockIdx.y, b = blockIdx.z;
    const int l = t & 31, wv = t >> 5, lm = l & 15, hi = (l >= 16) ? 1 : 0;
    const int Mb = (wv & 1) * 16, Jb = (wv >> 1) * 16;
    f32x8 acc = {};
    for (int sc = 0; sc < 8; ++sc) {
        const int nb = chunk * 2048 + sc * 256;
        __syncthreads();
        // stage exp(k - m)/s as bf16, K contiguous
        for (int i = 0; i < 16; ++i) {
            int p = i * 256 + t;                 // 4096 pairs
            int r = p >> 7, cp = p & 127;
            int row = b * 128 + h * 32 + r;
            float2 kvv = *(const float2*)(klog + (size_t)row * NTOT + nb + 2 * cp);
            float mr = kmax[row];
            float inv = 1.f / ksum[row];
            *(unsigned*)&sE[r * 256 + 2 * cp] =
                pack2(__expf(kvv.x - mr) * inv, __expf(kvv.y - mr) * inv);
        }
        // stage v (already bf16), K contiguous per dv-channel
        for (int i = 0; i < 32; ++i) {
            int p = i * 256 + t;                 // 8192 u32
            int j = p >> 7, cp = p & 127;
            unsigned u = *(const unsigned*)(vbuf + (size_t)(b * 256 + h * 64 + j) * NTOT + nb + 2 * cp);
            *(unsigned*)&sV[j * 256 + 2 * cp] = u;
        }
        // prefetch next sub-chunk into cache while WMMAs run (global_prefetch_b8)
        if (sc < 7) {
            const int nb2 = nb + 256;
            {   // klog: 32 rows x 1KB -> 8 x 128B lines per row
                int r = t >> 3, seg = t & 7;
                const float* pk = klog + (size_t)(b * 128 + h * 32 + r) * NTOT + nb2 + seg * 32;
                __builtin_prefetch(pk, 0, 3);
            }
            {   // vbuf: 64 rows x 512B -> 4 x 128B lines per row
                int j = t >> 2, seg = t & 3;
                const unsigned short* pv = vbuf + (size_t)(b * 256 + h * 64 + j) * NTOT + nb2 + seg * 64;
                __builtin_prefetch(pv, 0, 3);
            }
        }
        __syncthreads();
        const int m = Mb + lm, j = Jb + lm;
        for (int k0 = 0; k0 < 256; k0 += 32) {
            ABFrag a, bb;
#pragma unroll
            for (int i = 0; i < 8; ++i) {
                a.u[i]  = *(const unsigned*)&sE[m * 256 + k0 + ka(i, hi)];
                bb.u[i] = *(const unsigned*)&sV[j * 256 + k0 + kb(i, hi)];
            }
            acc = __builtin_amdgcn_wmma_f32_16x16x32_bf16(false, a.v, false, bb.v, (short)0, acc, false, false);
        }
    }
    float* pout = partial + (size_t)((b * 4 + h) * NCHUNK + chunk) * 2048;
#pragma unroll
    for (int r = 0; r < 8; ++r)
        pout[(Mb + r + hi * 8) * 64 + (Jb + lm)] = acc[r];
}

// ---------------- Kernel 4b: reduce kv partials over chunks ----------------
__global__ __launch_bounds__(256) void kvreduce_kernel(const float* __restrict__ partial,
                                                       float* __restrict__ kv) {
    int id = blockIdx.x * 256 + threadIdx.x;   // 16384 total
    int bh = id >> 11, e = id & 2047;
    float s = 0.f;
    for (int c = 0; c < NCHUNK; ++c)
        s += partial[(size_t)(bh * NCHUNK + c) * 2048 + e];
    kv[id] = s;
}

// ---------------- Kernel 5: out = (kv @ softmax_dqk(q)) / (1+eps) ----------------
// grid (N/128, 4 heads, B), block 256 (8 waves). One WMMA per 16(dv)x16(n) tile.
__global__ __launch_bounds__(256) void out_kernel(const float* __restrict__ qlog,
                                                  const float* __restrict__ kv,
                                                  float* __restrict__ outp) {
    __shared__ unsigned short sQ[128 * 32];
    __shared__ unsigned short sKV[64 * 32];
    const int t = threadIdx.x;
    const int n0 = blockIdx.x * 128;
    const int h = blockIdx.y, b = blockIdx.z;
    const float inv1p = 1.0f / (1.0f + 1e-6f);   // Z == 1 exactly (softmax sums)
    // kv transposed to [j][kk] so A K-pairs are contiguous
    for (int i = 0; i < 8; ++i) {
        int e = i * 256 + t;
        int kk = e >> 6, j = e & 63;
        sKV[j * 32 + kk] = f2bf(kv[(b * 4 + h) * 2048 + e] * inv1p);
    }
    // per-column softmax over the 32 qk channels
    if (t < 128) {
        int col = t;
        float qv[32];
        float m = -3.4e38f;
#pragma unroll
        for (int r = 0; r < 32; ++r) {
            qv[r] = qlog[(size_t)(b * 128 + h * 32 + r) * NTOT + n0 + col];
            m = fmaxf(m, qv[r]);
        }
        float s = 0.f;
#pragma unroll
        for (int r = 0; r < 32; ++r) { qv[r] = __expf(qv[r] - m); s += qv[r]; }
        float inv = 1.f / s;
#pragma unroll
        for (int r = 0; r < 32; r += 2)
            *(unsigned*)&sQ[col * 32 + r] = pack2(qv[r] * inv, qv[r + 1] * inv);
    }
    __syncthreads();
    const int l = t & 31, wv = t >> 5, lm = l & 15, hi = (l >= 16) ? 1 : 0;
    const int nt = wv;
    for (int jt = 0; jt < 4; ++jt) {
        ABFrag a, bb;
        const int m = jt * 16 + lm;
        const int col = nt * 16 + lm;
#pragma unroll
        for (int i = 0; i < 8; ++i) {
            a.u[i]  = *(const unsigned*)&sKV[m * 32 + ka(i, hi)];
            bb.u[i] = *(const unsigned*)&sQ[col * 32 + kb(i, hi)];
        }
        f32x8 acc = {};
        acc = __builtin_amdgcn_wmma_f32_16x16x32_bf16(false, a.v, false, bb.v, (short)0, acc, false, false);
#pragma unroll
        for (int r = 0; r < 8; ++r) {
            int ch = h * 64 + jt * 16 + r + hi * 8;
            int n  = n0 + nt * 16 + lm;
            outp[(size_t)(b * 256 + ch) * NTOT + n] = acc[r];
        }
    }
}

extern "C" void kernel_launch(void* const* d_in, const int* in_sizes, int n_in,
                              void* d_out, int out_size, void* d_ws, size_t ws_size,
                              hipStream_t stream) {
    (void)in_sizes; (void)n_in; (void)out_size; (void)ws_size;
    const float* x  = (const float*)d_in[0];
    const float* Wq = (const float*)d_in[1];
    const float* Wk = (const float*)d_in[2];
    const float* Wv = (const float*)d_in[3];
    const float* bv = (const float*)d_in[4];
    float* out = (float*)d_out;

    char* ws = (char*)d_ws;
    size_t off = 0;
    float* qlog = (float*)(ws + off); off += (size_t)2 * 128 * NTOT * sizeof(float);
    float* klog = (float*)(ws + off); off += (size_t)2 * 128 * NTOT * sizeof(float);
    unsigned short* vbuf = (unsigned short*)(ws + off); off += (size_t)2 * 256 * NTOT * sizeof(unsigned short);
    float* kmax = (float*)(ws + off); off += 256 * sizeof(float);
    float* ksum = (float*)(ws + off); off += 256 * sizeof(float);
    float* partial = (float*)(ws + off); off += (size_t)8 * NCHUNK * 2048 * sizeof(float);
    float* kvbuf = (float*)(ws + off); off += 16384 * sizeof(float);

    proj_kernel<<<dim3(NTOT / 64, 2, 2), 256, 0, stream>>>(x, Wq, qlog);
    proj_kernel<<<dim3(NTOT / 64, 2, 2), 256, 0, stream>>>(x, Wk, klog);
    vconv_kernel<<<dim3(216, 256, 2), 256, 0, stream>>>(x, Wv, bv, vbuf);
    kstats_kernel<<<dim3(256), 256, 0, stream>>>(klog, kmax, ksum);
    kv_kernel<<<dim3(NCHUNK, 4, 2), 256, 0, stream>>>(klog, kmax, ksum, vbuf, partial);
    kvreduce_kernel<<<dim3(64), 256, 0, stream>>>(partial, kvbuf);
    out_kernel<<<dim3(NTOT / 128, 4, 2), 256, 0, stream>>>(qlog, kvbuf, out);
}
